// Res_GCN_45638322487375
// MI455X (gfx1250) — compile-verified, matrix operand order checked by reading, not run on hardware
//
#include <hip/hip_runtime.h>

#define NN  100000
#define NNP 100096            // padded row count (multiple of 128)
#define NE  1600000
#define DD  128

typedef __attribute__((ext_vector_type(16))) __bf16 v16bf;
typedef __attribute__((ext_vector_type(8))) float v8f;
typedef __attribute__((ext_vector_type(4))) int v4i;

__device__ inline unsigned short f2bf(float f) {
    unsigned int u = __float_as_uint(f);
    u += 0x7fffu + ((u >> 16) & 1u);   // round-to-nearest-even
    return (unsigned short)(u >> 16);
}

union BFrag { unsigned int u[8]; v16bf v; };

// ---------------------------------------------------------------------------
// Async global->LDS copy (16 B per lane). Guarded: falls back to plain copy.
// Builtin prototype (from hipcc diagnostic):
//   void (v4i addrspace(1)* src_global, v4i addrspace(3)* dst_lds,
//         imm int offset, imm int cpol)
// ---------------------------------------------------------------------------
#if defined(__has_builtin)
#if __has_builtin(__builtin_amdgcn_global_load_async_to_lds_b128)
#define HAVE_ASYNC_LDS 1
#endif
#endif
#ifndef HAVE_ASYNC_LDS
#define HAVE_ASYNC_LDS 0
#endif

__device__ inline void async_copy16(const void* g, void* l) {
#if HAVE_ASYNC_LDS
    void* gm = const_cast<void*>(g);
    __builtin_amdgcn_global_load_async_to_lds_b128(
        (__attribute__((address_space(1))) v4i*)gm,
        (__attribute__((address_space(3))) v4i*)l, 0, 0);
#else
    *(uint4*)l = *(const uint4*)g;
#endif
}

__device__ inline void async_wait0() {
#if HAVE_ASYNC_LDS
#if __has_builtin(__builtin_amdgcn_s_wait_asynccnt)
    __builtin_amdgcn_s_wait_asynccnt(0);
#else
    asm volatile("s_wait_asynccnt 0x0" ::: "memory");
#endif
#endif
}

// ---------------------------------------------------------------------------
// agg = x  (the (1+eps)*x term with eps=0)
// ---------------------------------------------------------------------------
__global__ void copy_rows(const float* __restrict__ in, float* __restrict__ out, int n4) {
    int i = blockIdx.x * blockDim.x + threadIdx.x;
    if (i < n4) ((float4*)out)[i] = ((const float4*)in)[i];
}

// ---------------------------------------------------------------------------
// One-time weight conversion: wt[n*128+k] = bf16(W[k*128+n])   (16384 elems)
// ---------------------------------------------------------------------------
__global__ void convert_w(const float* __restrict__ W, unsigned short* __restrict__ wt) {
    int f = blockIdx.x * 256 + threadIdx.x;
    int k = f >> 7, n = f & 127;
    wt[n * DD + k] = f2bf(W[f]);
}

// ---------------------------------------------------------------------------
// agg[dst[e]] += x[src[e]]  — one edge per wave32, float4 per lane; x and agg
// (51.2 MB each) are L2-resident (192 MB), so gather + atomics run at L2 speed.
// ---------------------------------------------------------------------------
__global__ __launch_bounds__(256)
void scatter_add(const float* __restrict__ x, const int* __restrict__ src,
                 const int* __restrict__ dst, float* __restrict__ agg) {
    int e = blockIdx.x * 8 + (threadIdx.x >> 5);
    if (e >= NE) return;
    int lane = threadIdx.x & 31;
    int s = src[e];
    int d = dst[e];
    const float4 v = *(const float4*)(x + (size_t)s * DD + lane * 4);
    float* p = agg + (size_t)d * DD + lane * 4;
    atomicAdd(p + 0, v.x);
    atomicAdd(p + 1, v.y);
    atomicAdd(p + 2, v.z);
    atomicAdd(p + 3, v.w);
}

// ---------------------------------------------------------------------------
// out[M x 128] = relu(A[M x 128] @ W[128 x 128] + bias)
// bf16 WMMA, f32 accumulate. 256 threads = 8 waves; block = 128 rows,
// wave = 16 rows x 128 cols (8 accumulators, 4 k-steps of 32).
// W arrives pre-converted/transposed bf16 -> staged with async b128 copies.
// IN_BF16:  A is bf16 row-major  -> staged with async b128 copies (padded buf).
// !IN_BF16: A is f32 -> converted to bf16 during staging (bounds-checked).
// OUT_BF16: store bf16 (hidden h), else f32.
// ---------------------------------------------------------------------------
template <bool IN_BF16, bool OUT_BF16>
__global__ __launch_bounds__(256)
void gemm_bias_relu(const void* __restrict__ Ain, const unsigned short* __restrict__ Wt,
                    const float* __restrict__ bias, void* __restrict__ outp, int M) {
    __shared__ unsigned short sW[DD * DD];  // sW[n*128+k]
    __shared__ unsigned short sA[DD * DD];  // sA[r*128+k]

    const int t = threadIdx.x;
    const int blockRow = blockIdx.x * DD;

    // stage W: pure 32 KB copy (2048 x 16 B)
    for (int i = 0; i < 8; ++i) {
        int j = i * 256 + t;
        async_copy16((const char*)Wt + (size_t)j * 16, (char*)sW + (size_t)j * 16);
    }
    if (IN_BF16) {
        const char* Abase = (const char*)Ain + (size_t)blockRow * (DD * 2);
        for (int i = 0; i < 8; ++i) {
            int j = i * 256 + t;
            async_copy16(Abase + (size_t)j * 16, (char*)sA + (size_t)j * 16);
        }
    } else {
        const float* A = (const float*)Ain;
        for (int i = 0; i < 64; ++i) {
            int f = i * 256 + t;
            int r = f >> 7, k = f & 127;
            int row = blockRow + r;
            float v = (row < M) ? A[(size_t)row * DD + k] : 0.0f;
            sA[r * DD + k] = f2bf(v);
        }
    }
    async_wait0();
    __syncthreads();

    const int wave = t >> 5;
    const int lane = t & 31;
    const int m    = lane & 15;
    const int half = lane >> 4;
    const int r0   = wave * 16;

    v8f acc[8];
    const v8f zero = {0.f, 0.f, 0.f, 0.f, 0.f, 0.f, 0.f, 0.f};
#pragma unroll
    for (int nb = 0; nb < 8; ++nb) acc[nb] = zero;

#pragma unroll
    for (int kb = 0; kb < 4; ++kb) {
        const int k0 = kb * 32 + half * 8;
        BFrag af;
#pragma unroll
        for (int v = 0; v < 8; ++v) {
            int koff = k0 + ((v >= 4) ? 16 : 0) + (v & 3) * 2;
            af.u[v] = *(const unsigned int*)&sA[(r0 + m) * DD + koff];
        }
#pragma unroll
        for (int nb = 0; nb < 8; ++nb) {
            BFrag bfrag;
            const int n = nb * 16 + m;
#pragma unroll
            for (int v = 0; v < 8; ++v) {
                int koff = k0 + ((v >= 4) ? 16 : 0) + (v & 3) * 2;
                bfrag.u[v] = *(const unsigned int*)&sW[n * DD + koff];
            }
            acc[nb] = __builtin_amdgcn_wmma_f32_16x16x32_bf16(
                false, af.v, false, bfrag.v, (short)0, acc[nb], false, false);
        }
    }

#pragma unroll
    for (int nb = 0; nb < 8; ++nb) {
        const int n = nb * 16 + m;
        const float bv = bias[n];
#pragma unroll
        for (int r = 0; r < 8; ++r) {
            int row = blockRow + r0 + half * 8 + r;
            if (row < M) {
                float v = acc[nb][r] + bv;
                v = v > 0.0f ? v : 0.0f;
                if (OUT_BF16)
                    ((unsigned short*)outp)[(size_t)row * DD + n] = f2bf(v);
                else
                    ((float*)outp)[(size_t)row * DD + n] = v;
            }
        }
    }
}

// ---------------------------------------------------------------------------
extern "C" void kernel_launch(void* const* d_in, const int* in_sizes, int n_in,
                              void* d_out, int out_size, void* d_ws, size_t ws_size,
                              hipStream_t stream) {
    const float* x   = (const float*)d_in[0];
    const int*   ei  = (const int*)d_in[1];
    const int*   src = ei;
    const int*   dst = ei + NE;

    // per-layer weights in input order
    const float* Ws[6] = {(const float*)d_in[2],  (const float*)d_in[4],
                          (const float*)d_in[6],  (const float*)d_in[8],
                          (const float*)d_in[10], (const float*)d_in[12]};
    const float* b1[3] = {(const float*)d_in[3], (const float*)d_in[7],  (const float*)d_in[11]};
    const float* b2[3] = {(const float*)d_in[5], (const float*)d_in[9],  (const float*)d_in[13]};

    // workspace layout
    float* bufA = (float*)d_ws;                                 // agg        (51.2 MB)
    float* bufB = bufA + (size_t)NNP * DD;                      // layer out  (51.2 MB)
    unsigned short* hBF = (unsigned short*)(bufB + (size_t)NNP * DD);  // bf16 h (25.6 MB, padded)
    unsigned short* wBF = hBF + (size_t)NNP * DD;               // 6 bf16 weights (192 KB)

    float* outF = (float*)d_out;

    const int n4 = NN * DD / 4;
    dim3 cgrd((n4 + 255) / 256);
    dim3 sgrd((NE + 7) / 8);
    dim3 ggrd((NN + 127) / 128);

    // one-time bf16 weight conversion (transposed)
    for (int i = 0; i < 6; ++i)
        convert_w<<<64, 256, 0, stream>>>(Ws[i], wBF + (size_t)i * DD * DD);

    const float* xl = x;  // current layer input
    for (int l = 0; l < 3; ++l) {
        const unsigned short* w1 = wBF + (size_t)(2 * l + 0) * DD * DD;
        const unsigned short* w2 = wBF + (size_t)(2 * l + 1) * DD * DD;
        float* outLayer = (l == 2) ? outF : bufB;

        copy_rows  <<<cgrd, 256, 0, stream>>>(xl, bufA, n4);
        scatter_add<<<sgrd, 256, 0, stream>>>(xl, src, dst, bufA);
        gemm_bias_relu<false, true><<<ggrd, 256, 0, stream>>>(bufA, w1, b1[l], hBF, NN);
        gemm_bias_relu<true, false><<<ggrd, 256, 0, stream>>>(hBF, w2, b2[l], outLayer, NN);
        xl = outLayer;
    }
}
